// SpeakerClipLoss_73495480369485
// MI455X (gfx1250) — matrix operand order repaired and checked by reading, not compiled, not run on hardware
//
#include <hip/hip_runtime.h>

// ---------------------------------------------------------------------------
// Speaker CLIP loss on gfx1250 (MI455X), WMMA f16 path.
//   loss_i = 1/temp + log(Z_i) - (1/temp)*(P_i - 1)/cnt_i
//   Z_i = sum_j exp((e_i.e_j - 1)/temp)   (max of row == diagonal == 1/temp)
//   P_i = sum_{label_j == label_i} e_i.e_j  (includes j==i, subtract 1)
// ---------------------------------------------------------------------------

#define BN 8192
#define DK 256
#define CSPLIT 4
#define ROWS_PER_WG 128  // 8 waves * 16 rows

typedef __attribute__((ext_vector_type(16))) _Float16 v16h;
typedef __attribute__((ext_vector_type(8)))  _Float16 v8h;
typedef __attribute__((ext_vector_type(8)))  float    v8f;

union V16U { v16h v; v8h h[2]; };

__global__ __launch_bounds__(1024) void zero_hist_kernel(int* __restrict__ hist) {
    hist[threadIdx.x] = 0;
}

// One wave per row: L2-normalize f32 -> f16, and build the label histogram.
__global__ __launch_bounds__(256) void norm_kernel(const float* __restrict__ x,
                                                   const int* __restrict__ labels,
                                                   _Float16* __restrict__ eh,
                                                   int* __restrict__ hist) {
    const int wave = threadIdx.x >> 5;
    const int lane = threadIdx.x & 31;
    const int row  = blockIdx.x * 8 + wave;

    const float* xr = x + (size_t)row * DK;
    float v[8];
    float ss = 0.0f;
#pragma unroll
    for (int t = 0; t < 8; ++t) {
        float f = xr[lane + 32 * t];
        v[t] = f;
        ss += f * f;
    }
#pragma unroll
    for (int m = 16; m >= 1; m >>= 1) ss += __shfl_xor(ss, m);

    const float scale = 1.0f / fmaxf(sqrtf(ss), 1e-12f);
    _Float16* er = eh + (size_t)row * DK;
#pragma unroll
    for (int t = 0; t < 8; ++t) er[lane + 32 * t] = (_Float16)(v[t] * scale);

    if (lane == 0) atomicAdd(&hist[labels[row]], 1);
}

// Each wave: 16 rows (A resident in regs), streams 16-col B tiles over its
// column slice. 8x v_wmma_f32_16x16x32_f16 per tile, then exp + label-masked
// accumulation on the f32 accumulator fragment.
__global__ __launch_bounds__(256) void gemm_kernel(const _Float16* __restrict__ eh,
                                                   const int* __restrict__ labels,
                                                   const float* __restrict__ logt,
                                                   float* __restrict__ Zpart,
                                                   float* __restrict__ Ppart) {
    const int wave = threadIdx.x >> 5;
    const int lane = threadIdx.x & 31;
    const int half = lane >> 4;   // 0 or 1
    const int ln   = lane & 15;
    const int i0   = blockIdx.x * ROWS_PER_WG + wave * 16;
    const int cs   = blockIdx.y;
    const float inv_temp = __expf(-logt[0]);

    // A fragment: rows i0..i0+15, full K=256 -> 8 chunks of v16h.
    // 16-bit A 16x32 layout: half0 lane holds K {kk+0..7, kk+16..23},
    //                        half1 lane holds K {kk+8..15, kk+24..31}.
    v16h afrag[8];
    const _Float16* arow = eh + (size_t)(i0 + ln) * DK;
#pragma unroll
    for (int kc = 0; kc < 8; ++kc) {
        const int kk = kc * 32;
        V16U t;
        t.h[0] = *(const v8h*)(arow + kk + 8 * half);
        t.h[1] = *(const v8h*)(arow + kk + 16 + 8 * half);
        afrag[kc] = t.v;
    }

    // Labels of this lane's 8 accumulator rows (C layout: M = 8*half + r).
    int rlab[8];
#pragma unroll
    for (int r = 0; r < 8; ++r) rlab[r] = labels[i0 + 8 * half + r];

    v8f zsum = {0.f, 0.f, 0.f, 0.f, 0.f, 0.f, 0.f, 0.f};
    v8f psum = {0.f, 0.f, 0.f, 0.f, 0.f, 0.f, 0.f, 0.f};

    const int j_begin = cs * (BN / CSPLIT);
    const int j_end   = j_begin + (BN / CSPLIT);
    for (int j0 = j_begin; j0 < j_end; j0 += 16) {
        // B fragment: cols j0..j0+15 (rows of eh), 16-bit B 32x16 layout:
        // lane n holds K {kk+8*half..+7, kk+16+8*half..+7} (same as A).
        const _Float16* brow = eh + (size_t)(j0 + ln) * DK;
        const int clab = labels[j0 + ln];

        v8f acc = {0.f, 0.f, 0.f, 0.f, 0.f, 0.f, 0.f, 0.f};
#pragma unroll
        for (int kc = 0; kc < 8; ++kc) {
            const int kk = kc * 32;
            V16U t;
            t.h[0] = *(const v8h*)(brow + kk + 8 * half);
            t.h[1] = *(const v8h*)(brow + kk + 16 + 8 * half);
            acc = __builtin_amdgcn_wmma_f32_16x16x32_f16(
                false, afrag[kc], false, t.v, (short)0, acc, false, false);
        }
#pragma unroll
        for (int r = 0; r < 8; ++r) {
            const float s = acc[r];
            zsum[r] += __expf((s - 1.0f) * inv_temp);
            psum[r] += (clab == rlab[r]) ? s : 0.0f;
        }
    }

    // Reduce over the 16 lanes (distinct N) sharing each row set.
#pragma unroll
    for (int m = 8; m >= 1; m >>= 1) {
#pragma unroll
        for (int r = 0; r < 8; ++r) {
            zsum[r] += __shfl_xor(zsum[r], m);
            psum[r] += __shfl_xor(psum[r], m);
        }
    }
    if (ln == 0) {
#pragma unroll
        for (int r = 0; r < 8; ++r) {
            const int row = i0 + 8 * half + r;
            Zpart[cs * BN + row] = zsum[r];
            Ppart[cs * BN + row] = psum[r];
        }
    }
}

__global__ __launch_bounds__(1024) void final_kernel(const float* __restrict__ Zpart,
                                                     const float* __restrict__ Ppart,
                                                     const int* __restrict__ hist,
                                                     const int* __restrict__ labels,
                                                     const float* __restrict__ logt,
                                                     float* __restrict__ out) {
    __shared__ float s_tot[1024];
    __shared__ int   s_cnt[1024];
    const float inv_temp = __expf(-logt[0]);

    float tot = 0.0f;
    int   nv  = 0;
    for (int row = threadIdx.x; row < BN; row += 1024) {
        float Z = 0.0f, P = 0.0f;
#pragma unroll
        for (int cs = 0; cs < CSPLIT; ++cs) {
            Z += Zpart[cs * BN + row];
            P += Ppart[cs * BN + row];
        }
        const int cnt = hist[labels[row]] - 1;
        if (cnt > 0) {
            const float spos = (P - 1.0f) * inv_temp;  // drop the j==i term
            tot += inv_temp + logf(Z) - spos / (float)cnt;
            nv  += 1;
        }
    }
    s_tot[threadIdx.x] = tot;
    s_cnt[threadIdx.x] = nv;
    __syncthreads();
    for (int st = 512; st >= 1; st >>= 1) {
        if (threadIdx.x < st) {
            s_tot[threadIdx.x] += s_tot[threadIdx.x + st];
            s_cnt[threadIdx.x] += s_cnt[threadIdx.x + st];
        }
        __syncthreads();
    }
    if (threadIdx.x == 0)
        out[0] = (s_cnt[0] > 0) ? s_tot[0] / (float)s_cnt[0] : 0.0f;
}

extern "C" void kernel_launch(void* const* d_in, const int* in_sizes, int n_in,
                              void* d_out, int out_size, void* d_ws, size_t ws_size,
                              hipStream_t stream) {
    (void)in_sizes; (void)n_in; (void)out_size; (void)ws_size;
    const float* embeds = (const float*)d_in[0];
    const int*   labels = (const int*)d_in[1];
    const float* logt   = (const float*)d_in[2];
    float*       out    = (float*)d_out;

    char* ws = (char*)d_ws;
    _Float16* eh    = (_Float16*)ws;                                   // 4 MB
    float*    Zpart = (float*)(ws + (size_t)BN * DK * 2);              // 128 KB
    float*    Ppart = (float*)(ws + (size_t)BN * DK * 2 + (size_t)CSPLIT * BN * 4);
    int*      hist  = (int*)(ws + (size_t)BN * DK * 2 + (size_t)2 * CSPLIT * BN * 4);

    zero_hist_kernel<<<dim3(1), dim3(1024), 0, stream>>>(hist);
    norm_kernel<<<dim3(BN / 8), dim3(256), 0, stream>>>(embeds, labels, eh, hist);
    gemm_kernel<<<dim3(BN / ROWS_PER_WG, CSPLIT), dim3(256), 0, stream>>>(eh, labels, logt, Zpart, Ppart);
    final_kernel<<<dim3(1), dim3(1024), 0, stream>>>(Zpart, Ppart, hist, labels, logt, out);
}
